// LuminarkBeast_58480274702990
// MI455X (gfx1250) — matrix operand order repaired and checked by reading, not compiled, not run on hardware
//
#include <hip/hip_runtime.h>
#include <math.h>

// ---------------- problem constants (fixed by the reference) ----------------
constexpr int Vv   = 32000;
constexpr int Dm   = 1024;
constexpr int Ll   = 6;
constexpr int Hh   = 4;
constexpr int WINc = 5;
constexpr int Bb   = 2;
constexpr int Ss   = 2048;
constexpr int Mm   = Bb * Ss;      // 4096 token rows
constexpr int DFF  = 4 * Dm;       // 4096
constexpr int DH   = Dm / Hh;      // 256

constexpr int BM = 128, BN = 128, BK = 32;

// ---------------- types ----------------
typedef __attribute__((ext_vector_type(16))) __bf16    v16bf;
typedef __attribute__((ext_vector_type(8)))  float     v8f;
typedef __attribute__((ext_vector_type(4)))  unsigned  v4u;
typedef __attribute__((ext_vector_type(8)))  int       v8i;
typedef __attribute__((ext_vector_type(4)))  int       v4i;

struct __align__(16) U128    { unsigned x, y, z, w; };
struct __align__(16) Frag256 { U128 lo, hi; };           // 32B = 16 bf16
struct __align__(8)  US4     { unsigned short a, b, c, d; };

static __device__ __forceinline__ unsigned short f2bf(float f) {
  unsigned u = __builtin_bit_cast(unsigned, f);
  u += 0x7FFFu + ((u >> 16) & 1u);                       // round-to-nearest-even
  return (unsigned short)(u >> 16);
}

#if __has_builtin(__builtin_amdgcn_tensor_load_to_lds)
#define USE_TDM 1
#else
#define USE_TDM 0
#endif

#if USE_TDM
// Build a 2D-tile D# (ISA 08_async_tensor §8.3-8.6) and issue TENSOR_LOAD_TO_LDS.
// Tile: rows x 32 bf16 elements, row stride = K elements, packed contiguously in LDS.
// This toolchain exposes the 6-arg builtin: (g0 v4u, g1 v8i, g2 v4i, g3 v4i, v8i, cpol).
static __device__ __forceinline__ void tdm_load_tile(
    const unsigned short* gbase, unsigned ldsByteOff, int K, int rows)
{
  const unsigned long long ga = (unsigned long long)(size_t)gbase;
  v4u g0;
  g0.x = 1u;                                             // count=1 (valid user D#)
  g0.y = ldsByteOff;                                     // lds_addr
  g0.z = (unsigned)(ga & 0xFFFFFFFFu);                   // global_addr[31:0]
  g0.w = (unsigned)((ga >> 32) & 0x1FFFFFFu)             // global_addr[56:32]
       | (2u << 30);                                     // type=2 ("image")
  const unsigned td0 = (unsigned)K;                      // tensor_dim0
  const unsigned td1 = (unsigned)rows;                   // tensor_dim1
  const unsigned long long s0 = (unsigned long long)K;   // tensor_dim0_stride
  v8i g1;
  g1[0] = (int)(1u << 16);                               // data_size=1 -> 2 bytes
  g1[1] = (int)((td0 & 0xFFFFu) << 16);                  // tensor_dim0[15:0]
  g1[2] = (int)((td0 >> 16) | ((td1 & 0xFFFFu) << 16));  // dim0 hi | dim1 lo
  g1[3] = (int)((td1 >> 16) | ((unsigned)BK << 16));     // dim1 hi | tile_dim0=32
  g1[4] = (int)td1;                                      // tile_dim1=rows, tile_dim2=0
  g1[5] = (int)(unsigned)(s0 & 0xFFFFFFFFu);             // stride0[31:0]
  g1[6] = (int)(unsigned)((s0 >> 32) & 0xFFFFu);         // stride0[47:32], stride1=0
  g1[7] = 0;
  const v4i z4 = {0, 0, 0, 0};                           // groups 2/3 unused (2D)
  const v8i z8 = {0, 0, 0, 0, 0, 0, 0, 0};
  __builtin_amdgcn_tensor_load_to_lds(g0, g1, z4, z4, z8, 0);
}
#endif

// ---------------- tiled bf16 WMMA GEMM ----------------
// C[M,N] = A[M,K](bf16) x Bt[N,K](bf16, pre-transposed) + bias[N]  (+ epilogue)
// mode 0: plain   mode 1: += extra[m,n] (residual)   mode 2: *= sigmoid(extra[m,n])
// outBf: 0 -> f32 C, 1 -> bf16 C
__global__ __launch_bounds__(256) void gemm_bf16_wmma(
    const unsigned short* __restrict__ A, const unsigned short* __restrict__ Bt,
    const float* __restrict__ bias, const float* __restrict__ extra,
    void* __restrict__ Cout, int M, int N, int K, int mode, int outBf)
{
  __shared__ __align__(16) unsigned short sA[2][BM][BK];   // [m][k]  8KB/buf
  __shared__ __align__(16) unsigned short sB[2][BN][BK];   // [n][k]  8KB/buf

  const int tid   = threadIdx.x;
  const int lane  = tid & 31;
  const int wave  = tid >> 5;        // 8 waves
  const int wm    = wave >> 1;       // 0..3  (M)
  const int wn    = wave & 1;        // 0..1  (N)
  const int tileM = blockIdx.y * BM;
  const int tileN = blockIdx.x * BN;

  v8f acc[2][4];
  #pragma unroll
  for (int i = 0; i < 2; ++i)
    #pragma unroll
    for (int j = 0; j < 4; ++j) acc[i][j] = (v8f)0.0f;

  const int nk = K / BK;

#if USE_TDM
  const unsigned ldsAoff = (unsigned)(size_t)&sA[0][0][0];
  const unsigned ldsBoff = (unsigned)(size_t)&sB[0][0][0];
  auto stageTile = [&](int kt, int bufsel) {
    const int k0 = kt * BK;
    if (wave == 0)
      tdm_load_tile(&A[(size_t)tileM * K + k0], ldsAoff + (unsigned)bufsel * 8192u, K, BM);
    else if (wave == 1)
      tdm_load_tile(&Bt[(size_t)tileN * K + k0], ldsBoff + (unsigned)bufsel * 8192u, K, BN);
  };
#else
  auto stageTile = [&](int kt, int bufsel) {
    const int k0 = kt * BK;
    #pragma unroll
    for (int i = 0; i < 2; ++i) {                    // A: 512 uint4, 2/thread
      const int linear = tid + i * 256;
      const int row = linear >> 2;
      const int c8  = (linear & 3) << 3;
      *(U128*)&sA[bufsel][row][c8] =
          *(const U128*)&A[(size_t)(tileM + row) * K + k0 + c8];
    }
    #pragma unroll
    for (int i = 0; i < 2; ++i) {                    // B: 512 uint4, 2/thread
      const int linear = tid + i * 256;
      const int row = linear >> 2;
      const int c8  = (linear & 3) << 3;
      *(U128*)&sB[bufsel][row][c8] =
          *(const U128*)&Bt[(size_t)(tileN + row) * K + k0 + c8];
    }
  };
#endif

  int buf = 0;
  stageTile(0, 0);
#if USE_TDM
  if (wave < 2) __builtin_amdgcn_s_wait_tensorcnt(0);
#endif
  __syncthreads();

  for (int kt = 0; kt < nk; ++kt) {
    if (kt + 1 < nk) stageTile(kt + 1, buf ^ 1);

    const int kh  = (lane >> 4) << 3;   // A lane K-half: 0 or 8
    const int kb  = (lane >> 4) << 3;   // B lane K-half: 0 or 8
    const int l16 = lane & 15;

    #pragma unroll
    for (int mi = 0; mi < 2; ++mi) {
      const int ar = wm * 32 + mi * 16 + l16;
      Frag256 fa;
      fa.lo = *(const U128*)&sA[buf][ar][kh];
      fa.hi = *(const U128*)&sA[buf][ar][16 + kh];
      const v16bf a = __builtin_bit_cast(v16bf, fa);
      #pragma unroll
      for (int ni = 0; ni < 4; ++ni) {
        const int br = wn * 64 + ni * 16 + l16;
        Frag256 fb;
        fb.lo = *(const U128*)&sB[buf][br][kb];
        fb.hi = *(const U128*)&sB[buf][br][16 + kb];
        const v16bf b = __builtin_bit_cast(v16bf, fb);
        acc[mi][ni] = __builtin_amdgcn_wmma_f32_16x16x32_bf16(
            false, a, false, b, (short)0, acc[mi][ni], false, false);
      }
    }
#if USE_TDM
    if (wave < 2) __builtin_amdgcn_s_wait_tensorcnt(0);   // DMA overlapped compute
#endif
    __syncthreads();
    buf ^= 1;
  }

  // epilogue: VGPR r -> M = r + 8*(lane>=16), N = lane%16
  const int mh  = (lane >> 4) << 3;
  const int l16 = lane & 15;
  #pragma unroll
  for (int mi = 0; mi < 2; ++mi) {
    #pragma unroll
    for (int ni = 0; ni < 4; ++ni) {
      #pragma unroll
      for (int r = 0; r < 8; ++r) {
        const int gm = tileM + wm * 32 + mi * 16 + r + mh;
        const int gn = tileN + wn * 64 + ni * 16 + l16;
        float val = acc[mi][ni][r] + bias[gn];
        if (mode == 1) {
          val += extra[(size_t)gm * N + gn];
        } else if (mode == 2) {
          const float gate = extra[(size_t)gm * N + gn];
          val *= 1.0f / (1.0f + __expf(-gate));
        }
        if (outBf) ((unsigned short*)Cout)[(size_t)gm * N + gn] = f2bf(val);
        else       ((float*)Cout)[(size_t)gm * N + gn] = val;
      }
    }
  }
}

// ---------------- toroidal windowed attention: one wave per (b,s,head) ----------------
__global__ __launch_bounds__(256) void attn_win(
    const float* __restrict__ q, const float* __restrict__ k,
    const float* __restrict__ v, unsigned short* __restrict__ o)
{
  const int lane = threadIdx.x & 31;
  const int w    = blockIdx.x * 8 + (threadIdx.x >> 5);
  const int b    = w / (Ss * Hh);
  const int rem  = w % (Ss * Hh);
  const int s    = rem / Hh;
  const int hh   = rem % Hh;
  const size_t rowBase = ((size_t)(b * Ss + s)) * Dm + (size_t)hh * DH;

  float qv[8];
  #pragma unroll
  for (int j = 0; j < 8; ++j) qv[j] = q[rowBase + lane * 8 + j];

  const float scale = 0.0625f;                 // 1/sqrt(256)
  float sc[2 * WINc + 1];
  int   ssv[2 * WINc + 1];

  #pragma unroll
  for (int t = 0; t < 2 * WINc + 1; ++t) {
    int ss = s + t - WINc;
    ss = (ss % Ss + Ss) % Ss;
    ssv[t] = ss;
    const size_t kbase = ((size_t)(b * Ss + ss)) * Dm + (size_t)hh * DH;
    float p = 0.f;
    #pragma unroll
    for (int j = 0; j < 8; ++j) p += qv[j] * k[kbase + lane * 8 + j];
    #pragma unroll
    for (int off = 16; off > 0; off >>= 1) p += __shfl_xor(p, off, 32);
    sc[t] = p * scale;
  }

  float mx = sc[0];
  #pragma unroll
  for (int t = 1; t < 2 * WINc + 1; ++t) mx = fmaxf(mx, sc[t]);
  float sum = 0.f;
  #pragma unroll
  for (int t = 0; t < 2 * WINc + 1; ++t) { sc[t] = __expf(sc[t] - mx); sum += sc[t]; }
  const float inv = 1.0f / sum;

  float accv[8] = {0.f, 0.f, 0.f, 0.f, 0.f, 0.f, 0.f, 0.f};
  #pragma unroll
  for (int t = 0; t < 2 * WINc + 1; ++t) {
    const size_t vbase = ((size_t)(b * Ss + ssv[t])) * Dm + (size_t)hh * DH;
    const float wt = sc[t] * inv;
    #pragma unroll
    for (int j = 0; j < 8; ++j) accv[j] += wt * v[vbase + lane * 8 + j];
  }
  US4 p0, p1;
  p0.a = f2bf(accv[0]); p0.b = f2bf(accv[1]); p0.c = f2bf(accv[2]); p0.d = f2bf(accv[3]);
  p1.a = f2bf(accv[4]); p1.b = f2bf(accv[5]); p1.c = f2bf(accv[6]); p1.d = f2bf(accv[7]);
  *(US4*)&o[rowBase + lane * 8]     = p0;
  *(US4*)&o[rowBase + lane * 8 + 4] = p1;
}

// ---------------- embedding gather (f32) ----------------
__global__ __launch_bounds__(256) void embed_kernel(
    const int* __restrict__ x, const float* __restrict__ emb, float* __restrict__ h)
{
  const size_t i4 = (size_t)blockIdx.x * blockDim.x + threadIdx.x;  // float4 index
  const int    m  = (int)(i4 / (Dm / 4));
  const int    d  = (int)(i4 % (Dm / 4)) * 4;
  const int    row = x[m];
  *(float4*)&h[(size_t)m * Dm + d] = *(const float4*)&emb[(size_t)row * Dm + d];
}

// ---------------- f32 -> bf16 (8 elems/thread) ----------------
__global__ __launch_bounds__(256) void cvt_bf16_kernel(
    const float* __restrict__ src, unsigned short* __restrict__ dst)
{
  const size_t i = ((size_t)blockIdx.x * blockDim.x + threadIdx.x) * 8;
  const float4 a = *(const float4*)&src[i];
  const float4 b = *(const float4*)&src[i + 4];
  US4 p0, p1;
  p0.a = f2bf(a.x); p0.b = f2bf(a.y); p0.c = f2bf(a.z); p0.d = f2bf(a.w);
  p1.a = f2bf(b.x); p1.b = f2bf(b.y); p1.c = f2bf(b.z); p1.d = f2bf(b.w);
  *(US4*)&dst[i]     = p0;
  *(US4*)&dst[i + 4] = p1;
}

// ---------------- f32 KxN -> bf16 NxK transpose (32x32 LDS tiles) ----------------
__global__ __launch_bounds__(256) void transpose_bf16_kernel(
    const float* __restrict__ src, unsigned short* __restrict__ dst, int K, int N)
{
  __shared__ float tile[32][33];
  const int tx = threadIdx.x, ty = threadIdx.y;       // (32, 8)
  const int bx = blockIdx.x,  by = blockIdx.y;        // (N/32, K/32)
  #pragma unroll
  for (int i = 0; i < 4; ++i) {
    const int r = ty + i * 8;                         // k within tile
    tile[r][tx] = src[(size_t)(by * 32 + r) * N + bx * 32 + tx];
  }
  __syncthreads();
  #pragma unroll
  for (int i = 0; i < 4; ++i) {
    const int r = ty + i * 8;                         // n within tile
    dst[(size_t)(bx * 32 + r) * K + by * 32 + tx] = f2bf(tile[tx][r]);
  }
}

// ---------------- driver ----------------
extern "C" void kernel_launch(void* const* d_in, const int* in_sizes, int n_in,
                              void* d_out, int out_size, void* d_ws, size_t ws_size,
                              hipStream_t stream) {
  (void)in_sizes; (void)n_in; (void)out_size; (void)ws_size;
  const int*   x   = (const int*)  d_in[0];
  const float* emb = (const float*)d_in[1];
  const float* Wq  = (const float*)d_in[2];
  const float* bq  = (const float*)d_in[3];
  const float* Wk  = (const float*)d_in[4];
  const float* bk  = (const float*)d_in[5];
  const float* Wv  = (const float*)d_in[6];
  const float* bv  = (const float*)d_in[7];
  const float* Wo  = (const float*)d_in[8];
  const float* bo  = (const float*)d_in[9];
  const float* Wg  = (const float*)d_in[10];
  const float* bg  = (const float*)d_in[11];
  const float* Wu  = (const float*)d_in[12];
  const float* bu  = (const float*)d_in[13];
  const float* W2  = (const float*)d_in[14];
  const float* b2  = (const float*)d_in[15];
  const float* Wh  = (const float*)d_in[16];
  const float* bh  = (const float*)d_in[17];
  float* out = (float*)d_out;

  // ---- workspace layout ----
  // f32:  h [Mm*Dm] | R [Mm*DFF]  (R holds q|k|v during attention, g during MLP)
  // bf16: hb | ob | ub | Wqt..Wot | Wgt | Wut | W2t | Wht
  float* h = (float*)d_ws;
  float* R = h + (size_t)Mm * Dm;
  float* q = R;
  float* k = R + (size_t)Mm * Dm;
  float* v = R + 2 * (size_t)Mm * Dm;
  float* g = R;
  unsigned short* us = (unsigned short*)(R + (size_t)Mm * DFF);
  unsigned short* hb  = us; us += (size_t)Mm * Dm;
  unsigned short* ob  = us; us += (size_t)Mm * Dm;
  unsigned short* ub  = us; us += (size_t)Mm * DFF;
  unsigned short* Wqt = us; us += (size_t)Ll * Dm * Dm;
  unsigned short* Wkt = us; us += (size_t)Ll * Dm * Dm;
  unsigned short* Wvt = us; us += (size_t)Ll * Dm * Dm;
  unsigned short* Wot = us; us += (size_t)Ll * Dm * Dm;
  unsigned short* Wgt = us; us += (size_t)Dm * DFF;
  unsigned short* Wut = us; us += (size_t)Dm * DFF;
  unsigned short* W2t = us; us += (size_t)Dm * DFF;
  unsigned short* Wht = us; /* Vv*Dm */

  const dim3 blk(256);
  const dim3 tb(32, 8);
  const dim3 gD(Dm / BN,  Mm / BM);
  const dim3 gF(DFF / BN, Mm / BM);
  const dim3 gV(Vv / BN,  Mm / BM);
  const int cvtBlocks = (Mm * Dm) / (256 * 8);

  // one-time weight convert+transpose (bf16, N-major for contiguous B tiles)
  for (int l = 0; l < Ll; ++l) {
    const size_t wo = (size_t)l * Dm * Dm;
    transpose_bf16_kernel<<<dim3(Dm/32, Dm/32), tb, 0, stream>>>(Wq + wo, Wqt + wo, Dm, Dm);
    transpose_bf16_kernel<<<dim3(Dm/32, Dm/32), tb, 0, stream>>>(Wk + wo, Wkt + wo, Dm, Dm);
    transpose_bf16_kernel<<<dim3(Dm/32, Dm/32), tb, 0, stream>>>(Wv + wo, Wvt + wo, Dm, Dm);
    transpose_bf16_kernel<<<dim3(Dm/32, Dm/32), tb, 0, stream>>>(Wo + wo, Wot + wo, Dm, Dm);
  }
  transpose_bf16_kernel<<<dim3(DFF/32, Dm/32), tb, 0, stream>>>(Wg, Wgt, Dm, DFF);
  transpose_bf16_kernel<<<dim3(DFF/32, Dm/32), tb, 0, stream>>>(Wu, Wut, Dm, DFF);
  transpose_bf16_kernel<<<dim3(Dm/32, DFF/32), tb, 0, stream>>>(W2, W2t, DFF, Dm);
  transpose_bf16_kernel<<<dim3(Vv/32, Dm/32),  tb, 0, stream>>>(Wh, Wht, Dm, Vv);

  embed_kernel<<<(Mm * Dm / 4) / 256, blk, 0, stream>>>(x, emb, h);
  cvt_bf16_kernel<<<cvtBlocks, blk, 0, stream>>>(h, hb);

  for (int l = 0; l < Ll; ++l) {
    const size_t wo = (size_t)l * Dm * Dm;
    const size_t bo_ = (size_t)l * Dm;
    gemm_bf16_wmma<<<gD, blk, 0, stream>>>(hb, Wqt + wo, bq + bo_, nullptr, q, Mm, Dm, Dm, 0, 0);
    gemm_bf16_wmma<<<gD, blk, 0, stream>>>(hb, Wkt + wo, bk + bo_, nullptr, k, Mm, Dm, Dm, 0, 0);
    gemm_bf16_wmma<<<gD, blk, 0, stream>>>(hb, Wvt + wo, bv + bo_, nullptr, v, Mm, Dm, Dm, 0, 0);
    attn_win<<<(Bb * Ss * Hh) / 8, blk, 0, stream>>>(q, k, v, ob);
    gemm_bf16_wmma<<<gD, blk, 0, stream>>>(ob, Wot + wo, bo + bo_, h, h, Mm, Dm, Dm, 1, 0);
    cvt_bf16_kernel<<<cvtBlocks, blk, 0, stream>>>(h, hb);
  }

  gemm_bf16_wmma<<<gF, blk, 0, stream>>>(hb, Wgt, bg, nullptr, g,   Mm, DFF, Dm, 0, 0);
  gemm_bf16_wmma<<<gF, blk, 0, stream>>>(hb, Wut, bu, g,       ub,  Mm, DFF, Dm, 2, 1);
  gemm_bf16_wmma<<<gD, blk, 0, stream>>>(ub, W2t, b2, h,       h,   Mm, Dm, DFF, 1, 0);
  cvt_bf16_kernel<<<cvtBlocks, blk, 0, stream>>>(h, hb);
  gemm_bf16_wmma<<<gV, blk, 0, stream>>>(hb, Wht, bh, nullptr, out, Mm, Vv, Dm, 0, 0);
}
